// BasicBlockIMCFlow_65300682768730
// MI455X (gfx1250) — 8-bit matrix operand order repaired and checked by reading, compiled, not run on hardware
//
#include <hip/hip_runtime.h>
#include <cstdint>

typedef int v8i __attribute__((ext_vector_type(8)));

#define Bn 64
#define Cn 64
#define Hn 64
#define Wn 64
#define HP 66
#define WP 66
#define NPIX (Bn * Cn * Hn * Wn)
#define ACT_PAD_BYTES ((size_t)Bn * HP * WP * 64)
#define WPACK_INTS (9 * 4 * 32 * 8)


__global__ __launch_bounds__(256) void zero_ws(uint4* __restrict__ p, int n16) {
  int i = blockIdx.x * 256 + threadIdx.x;
  if (i < n16) p[i] = make_uint4(0u, 0u, 0u, 0u);
}

__global__ __launch_bounds__(256) void quant_input(
    const float* __restrict__ x, const float* __restrict__ qmn,
    const float* __restrict__ qmx, short* __restrict__ xint,
    unsigned char* __restrict__ qout) {
  const int i = blockIdx.x * 256 + threadIdx.x;
  const float mn = *qmn;
  float step = floorf((*qmx - mn) * 0.0625f);
  const float inv = (step > 0.f) ? (1.f / step) : 1.f;
  float v = rintf(x[i] * 256.f);
  v = fminf(fmaxf(v, -32768.f), 32767.f);
  xint[i] = (short)v;
  float q = floorf((v - mn) * inv);
  q = fminf(fmaxf(q, 0.f), 15.f);
  qout[i] = (unsigned char)q;
}

__global__ __launch_bounds__(256) void nchw8_to_nhwc_pad(
    const unsigned char* __restrict__ qin, unsigned char* __restrict__ act) {
  const int idx = blockIdx.x * 256 + threadIdx.x;
  const int b = idx >> 12, y = (idx >> 6) & 63, x = idx & 63;
  const size_t in0 = (size_t)b * Cn * (Hn * Wn) + (size_t)y * Wn + x;
  unsigned d[16];
#pragma unroll
  for (int c4 = 0; c4 < 16; ++c4) {
    unsigned w = 0;
#pragma unroll
    for (int j = 0; j < 4; ++j)
      w |= (unsigned)qin[in0 + (size_t)(c4 * 4 + j) * (Hn * Wn)] << (8 * j);
    d[c4] = w;
  }
  uint4* op = (uint4*)(act + (((size_t)b * HP + (y + 1)) * WP + (x + 1)) * 64);
  op[0] = make_uint4(d[0], d[1], d[2], d[3]);
  op[1] = make_uint4(d[4], d[5], d[6], d[7]);
  op[2] = make_uint4(d[8], d[9], d[10], d[11]);
  op[3] = make_uint4(d[12], d[13], d[14], d[15]);
}

__global__ __launch_bounds__(128) void pack_weights(
    const float* __restrict__ w, int* __restrict__ wp) {
  const int t = blockIdx.x;
  const int nt = threadIdx.x >> 5;
  const int lane = threadIdx.x & 31;
  const int co = nt * 16 + (lane & 15);
  const int khalf = (lane >> 4) * 16;
#pragma unroll
  for (int v = 0; v < 8; ++v) {
    const int kbase = (v >> 1) * 16 + (lane >> 4) * 8 + (v & 1) * 4;
    unsigned wd = 0;
#pragma unroll
    for (int j = 0; j < 4; ++j) {
      const int ci = kbase + j;
      const int iv = (int)w[(size_t)(co * Cn + ci) * 9 + t];
      wd |= ((unsigned)iv & 0xFFu) << (8 * j);
    }
    wp[((t * 4 + nt) * 32 + lane) * 8 + v] = (int)wd;
  }
}


__device__ __forceinline__ v8i load_afrag(const unsigned char* p) {
  uint2 c0 = *(const uint2*)(p);
  uint2 c1 = *(const uint2*)(p + 16);
  uint2 c2 = *(const uint2*)(p + 32);
  uint2 c3 = *(const uint2*)(p + 48);
  v8i a;
  a[0] = (int)c0.x; a[1] = (int)c0.y;
  a[2] = (int)c1.x; a[3] = (int)c1.y;
  a[4] = (int)c2.x; a[5] = (int)c2.y;
  a[6] = (int)c3.x; a[7] = (int)c3.y;
  return a;
}

__device__ __forceinline__ void conv_mma(const unsigned char* __restrict__ act,
                                         const int* lds_w, int b, int y, int x0,
                                         int lane, v8i acc[4]) {
  const int m = lane & 15;
  const int hi8 = (lane >> 4) * 8;
  const unsigned char* __restrict__ base =
      act + (((size_t)b * HP + (size_t)y) * WP + (size_t)(x0 + m)) * 64 + hi8;
  const int lbase = lane * 8;
#pragma unroll
  for (int t = 0; t < 9; ++t) {
    const int ky = t / 3, kx = t % 3;
    const v8i a = load_afrag(base + (ky * WP + kx) * 64);
    const v8i b0 = *(const v8i*)&lds_w[(t * 4 + 0) * 256 + lbase];
    const v8i b1 = *(const v8i*)&lds_w[(t * 4 + 1) * 256 + lbase];
    const v8i b2 = *(const v8i*)&lds_w[(t * 4 + 2) * 256 + lbase];
    const v8i b3 = *(const v8i*)&lds_w[(t * 4 + 3) * 256 + lbase];
    acc[0] = __builtin_amdgcn_wmma_i32_16x16x64_iu8(false, a, true, b0, acc[0], false, false);
    acc[1] = __builtin_amdgcn_wmma_i32_16x16x64_iu8(false, a, true, b1, acc[1], false, false);
    acc[2] = __builtin_amdgcn_wmma_i32_16x16x64_iu8(false, a, true, b2, acc[2], false, false);
    acc[3] = __builtin_amdgcn_wmma_i32_16x16x64_iu8(false, a, true, b3, acc[3], false, false);
  }
}

__global__ __launch_bounds__(256) void conv_bn_quant(
    const unsigned char* __restrict__ act, const int* __restrict__ wpack,
    const float* __restrict__ bscale, const float* __restrict__ bbias,
    const float* __restrict__ qmn, const float* __restrict__ qmx,
    unsigned char* __restrict__ qout) {
  __shared__ int lds_w[WPACK_INTS];
  for (int i = threadIdx.x; i < WPACK_INTS; i += 256) lds_w[i] = wpack[i];
  __syncthreads();

  const int lane = threadIdx.x & 31;
  const int strip = blockIdx.x * 8 + (threadIdx.x >> 5);
  const int b = strip >> 8, sp = strip & 255;
  const int y = sp >> 2, x0 = (sp & 3) << 4;

  v8i acc[4] = {};
  conv_mma(act, lds_w, b, y, x0, lane, acc);

  const float mn = *qmn;
  float step = floorf((*qmx - mn) * 0.0625f);
  const float inv = (step > 0.f) ? (1.f / step) : 1.f;
  const int cl = lane & 15;
  const int xb = x0 + (lane >> 4) * 8;
#pragma unroll
  for (int nt = 0; nt < 4; ++nt) {
    const int c = nt * 16 + cl;
    const float sc = bscale[c], bi = bbias[c];
    unsigned lo = 0, hiw = 0;
#pragma unroll
    for (int r = 0; r < 8; ++r) {
      const float h = (float)acc[nt][r] * sc + bi;
      float q = fminf(fmaxf(floorf((h - mn) * inv), 0.f), 15.f);
      const unsigned qi = (unsigned)q & 0xFFu;
      if (r < 4) lo |= qi << (8 * r); else hiw |= qi << (8 * (r - 4));
    }
    const size_t o = (((size_t)b * Cn + c) * Hn + y) * Wn + xb;
    *(uint2*)(qout + o) = make_uint2(lo, hiw);
  }
}

__global__ __launch_bounds__(256) void conv_bn_res_out(
    const unsigned char* __restrict__ act, const int* __restrict__ wpack,
    const float* __restrict__ bscale, const float* __restrict__ bbias,
    const short* __restrict__ xint, float* __restrict__ out) {
  __shared__ int lds_w[WPACK_INTS];
  for (int i = threadIdx.x; i < WPACK_INTS; i += 256) lds_w[i] = wpack[i];
  __syncthreads();

  const int lane = threadIdx.x & 31;
  const int strip = blockIdx.x * 8 + (threadIdx.x >> 5);
  const int b = strip >> 8, sp = strip & 255;
  const int y = sp >> 2, x0 = (sp & 3) << 4;

  v8i acc[4] = {};
  conv_mma(act, lds_w, b, y, x0, lane, acc);

  const int cl = lane & 15;
  const int xb = x0 + (lane >> 4) * 8;
#pragma unroll
  for (int nt = 0; nt < 4; ++nt) {
    const int c = nt * 16 + cl;
    const float sc = bscale[c], bi = bbias[c];
    const size_t o = (((size_t)b * Cn + c) * Hn + y) * Wn + xb;
    const uint4 xr = *(const uint4*)(xint + o);
    const unsigned wds[4] = {xr.x, xr.y, xr.z, xr.w};
    float res[8];
#pragma unroll
    for (int r = 0; r < 8; ++r) {
      const short xi = (short)((wds[r >> 1] >> ((r & 1) * 16)) & 0xFFFFu);
      const float h = (float)acc[nt][r] * sc + bi + (float)xi;
      float v = rintf(h);
      v = fminf(fmaxf(v, -32768.f), 32767.f);
      res[r] = v * 0.00390625f;
    }
    *(float4*)(out + o)     = make_float4(res[0], res[1], res[2], res[3]);
    *(float4*)(out + o + 4) = make_float4(res[4], res[5], res[6], res[7]);
  }
}


extern "C" void kernel_launch(void* const* d_in, const int* in_sizes, int n_in,
                              void* d_out, int out_size, void* d_ws, size_t ws_size,
                              hipStream_t stream) {
  (void)in_sizes; (void)n_in; (void)out_size; (void)ws_size;
  const float* x    = (const float*)d_in[0];
  const float* w1   = (const float*)d_in[1];
  const float* bn1s = (const float*)d_in[2];
  const float* bn1b = (const float*)d_in[3];
  const float* w2   = (const float*)d_in[4];
  const float* bn2s = (const float*)d_in[5];
  const float* bn2b = (const float*)d_in[6];
  const float* a1mn = (const float*)d_in[7];
  const float* a1mx = (const float*)d_in[8];
  const float* a2mn = (const float*)d_in[9];
  const float* a2mx = (const float*)d_in[10];
  float* out = (float*)d_out;

  char* ws = (char*)d_ws;
  size_t off = 0;
  short* xint = (short*)(ws + off);                off += (size_t)NPIX * 2;
  unsigned char* act = (unsigned char*)(ws + off); off += ACT_PAD_BYTES;
  unsigned char* qbuf = (unsigned char*)(ws + off); off += (size_t)NPIX;
  int* wp1 = (int*)(ws + off);                     off += (size_t)WPACK_INTS * 4;
  int* wp2 = (int*)(ws + off);                     off += (size_t)WPACK_INTS * 4;

  const int n16 = (int)(ACT_PAD_BYTES / 16);
  zero_ws<<<(n16 + 255) / 256, 256, 0, stream>>>((uint4*)act, n16);

  pack_weights<<<9, 128, 0, stream>>>(w1, wp1);
  pack_weights<<<9, 128, 0, stream>>>(w2, wp2);

  quant_input<<<NPIX / 256, 256, 0, stream>>>(x, a1mn, a1mx, xint, qbuf);
  nchw8_to_nhwc_pad<<<(Bn * Hn * Wn) / 256, 256, 0, stream>>>(qbuf, act);
  conv_bn_quant<<<2048, 256, 0, stream>>>(act, wp1, bn1s, bn1b, a2mn, a2mx, qbuf);
  nchw8_to_nhwc_pad<<<(Bn * Hn * Wn) / 256, 256, 0, stream>>>(qbuf, act);
  conv_bn_res_out<<<2048, 256, 0, stream>>>(act, wp2, bn2s, bn2b, xint, out);
}
